// Scaled_Dot_Product_90623809945791
// MI455X (gfx1250) — matrix operand order repaired and checked by reading, not compiled, not run on hardware
//
#include <hip/hip_runtime.h>
#include <stdint.h>

typedef __attribute__((ext_vector_type(16))) _Float16 v16h;
typedef __attribute__((ext_vector_type(8)))  _Float16 v8h;
typedef __attribute__((ext_vector_type(2)))  _Float16 h2;
typedef __attribute__((ext_vector_type(2)))  __fp16   h2raw;
typedef __attribute__((ext_vector_type(8)))  float    v8f;

#define SEQ    2048
#define DH     64
#define BN     32
#define NITER  (SEQ / BN)
// softmax computed in exp2 domain: scores carry SCALE*log2e, mask adds -1e9*log2e
#define QSCALE (0.125f * 1.44269504088896f)
#define CMASK  (-1.0e9f * 1.44269504088896f)

__device__ __forceinline__ h2 pkrtz(float a, float b) {
  h2raw t = __builtin_amdgcn_cvt_pkrtz(a, b);   // v_cvt_pk_rtz_f16_f32
  return __builtin_bit_cast(h2, t);
}

// xor-16 lane swap as a single VALU op (v_permlanex16_b32), no LDS round-trip
__device__ __forceinline__ float xor16(float x) {
  int v = __builtin_bit_cast(int, x);
  int y = __builtin_amdgcn_permlanex16(v, v, 0x76543210, 0xfedcba98, false, false);
  return __builtin_bit_cast(float, y);
}

__device__ __forceinline__ v16h cat16(v8h lo, v8h hi) {
  v16h r;
#pragma unroll
  for (int i = 0; i < 8; ++i) { r[i] = lo[i]; r[i + 8] = hi[i]; }
  return r;
}

__device__ __forceinline__ v8h ld8(const _Float16* p) {
  return *(const v8h*)p;
}

__launch_bounds__(128)
__global__ void fa_fwd_kernel(const float* __restrict__ Q,
                              const float* __restrict__ K,
                              const float* __restrict__ V,
                              const unsigned char* __restrict__ M,
                              float* __restrict__ O)
{
  // K tile [key][d] f16, row padded to 72 halves (144B) -> 16B-aligned b128 reads
  __shared__ alignas(16) _Float16 ldsK[2][BN][72];
  // V tile transposed [d][key] f16, row padded to 40 halves (80B)
  __shared__ alignas(16) _Float16 ldsVT[2][DH][40];

  const int tid  = threadIdx.x;
  const int lane = tid & 31;
  const int wave = tid >> 5;
  const int hi   = lane >> 4;   // which 16-lane half
  const int ln   = lane & 15;

  const int bh = blockIdx.y;
  const int qb = blockIdx.x * 64 + wave * 16;   // 16 queries per wave

  const size_t bhSD = (size_t)bh * SEQ * DH;
  const float* Qrow = Q + bhSD + (size_t)(qb + ln) * DH;
  const float* Kb   = K + bhSD;
  const float* Vb   = V + bhSD;
  const unsigned char* Mrow = M + (size_t)bh * SEQ * SEQ + (size_t)(qb + ln) * SEQ;
  float* Orow = O + bhSD + (size_t)(qb + ln) * DH;

  // ---- Q in B-matrix layout (once), with SCALE*log2e folded into the f16 ----
  const int klo = hi * 16;
  v16h bq0, bq1;
#pragma unroll
  for (int e = 0; e < 16; e += 2) {
    h2 t0 = pkrtz(Qrow[klo + e] * QSCALE, Qrow[klo + e + 1] * QSCALE);
    h2 t1 = pkrtz(Qrow[32 + klo + e] * QSCALE, Qrow[32 + klo + e + 1] * QSCALE);
    bq0[e] = t0[0]; bq0[e + 1] = t0[1];
    bq1[e] = t1[0]; bq1[e + 1] = t1[1];
  }

  // all-ones A matrix: one extra WMMA computes the softmax row-sum (over all 32
  // keys of the block, both lane-halves) entirely in the matrix pipe
  v16h aone;
#pragma unroll
  for (int i = 0; i < 16; ++i) aone[i] = (_Float16)1.0f;

  // cooperative staging indices
  const int skey = tid & 31;          // K staging: key within tile
  const int sd0  = (tid >> 5) * 16;   // K staging: d base
  const int vkp  = (tid & 15) * 2;    // V staging: key pair
  const int vd0  = (tid >> 4) * 8;    // V staging: d base

  float4 kf0, kf1, kf2, kf3;          // prefetch regs: K
  float4 va0, va1, vb0, vb1;          // prefetch regs: V
  unsigned long long mm0, mm1;        // prefetch regs: mask (8B per score tile)

  auto loadStage = [&](int it) {
    const float* kp = Kb + (size_t)(it * BN + skey) * DH + sd0;
    kf0 = ((const float4*)kp)[0];
    kf1 = ((const float4*)kp)[1];
    kf2 = ((const float4*)kp)[2];
    kf3 = ((const float4*)kp)[3];
    const float* vp = Vb + (size_t)(it * BN + vkp) * DH + vd0;
    va0 = ((const float4*)vp)[0];
    va1 = ((const float4*)vp)[1];
    const float* vq = vp + DH;
    vb0 = ((const float4*)vq)[0];
    vb1 = ((const float4*)vq)[1];
    mm0 = __builtin_nontemporal_load(
        (const unsigned long long*)(Mrow + it * BN + hi * 8));
    mm1 = __builtin_nontemporal_load(
        (const unsigned long long*)(Mrow + it * BN + 16 + hi * 8));
  };

  auto storeStage = [&](int buf) {
    // K: [key][d] contiguous f16, two b128 stores
    _Float16* kd = &ldsK[buf][skey][sd0];
    h2 k0 = pkrtz(kf0.x, kf0.y);
    h2 k1 = pkrtz(kf0.z, kf0.w);
    h2 k2 = pkrtz(kf1.x, kf1.y);
    h2 k3 = pkrtz(kf1.z, kf1.w);
    h2 k4 = pkrtz(kf2.x, kf2.y);
    h2 k5 = pkrtz(kf2.z, kf2.w);
    h2 k6 = pkrtz(kf3.x, kf3.y);
    h2 k7 = pkrtz(kf3.z, kf3.w);
    v8h h0, h1;
    h0[0]=k0[0]; h0[1]=k0[1]; h0[2]=k1[0]; h0[3]=k1[1];
    h0[4]=k2[0]; h0[5]=k2[1]; h0[6]=k3[0]; h0[7]=k3[1];
    h1[0]=k4[0]; h1[1]=k4[1]; h1[2]=k5[0]; h1[3]=k5[1];
    h1[4]=k6[0]; h1[5]=k6[1]; h1[6]=k7[0]; h1[7]=k7[1];
    *(v8h*)kd       = h0;
    *(v8h*)(kd + 8) = h1;
    // V transposed: pack a key-pair per 4B store (conflict-free b32 stores)
    float av[8] = {va0.x, va0.y, va0.z, va0.w, va1.x, va1.y, va1.z, va1.w};
    float bv[8] = {vb0.x, vb0.y, vb0.z, vb0.w, vb1.x, vb1.y, vb1.z, vb1.w};
#pragma unroll
    for (int i = 0; i < 8; ++i) {
      h2 t = pkrtz(av[i], bv[i]);
      *(h2*)&ldsVT[buf][vd0 + i][vkp] = t;
    }
  };

  v8f zz = {};
  v8f acc[4]; acc[0] = zz; acc[1] = zz; acc[2] = zz; acc[3] = zz;
  float m_run = -__builtin_inff();
  float l_run = 0.0f;

  loadStage(0);

#pragma unroll 2
  for (int it = 0; it < NITER; ++it) {
    const int buf = it & 1;
    storeStage(buf);
    __syncthreads();

    const unsigned long long cm0 = mm0, cm1 = mm1;
    if (it + 1 < NITER) loadStage(it + 1);   // pipeline next KV block

    // ---- mask -> WMMA C operand: C[r] = byte * (-1e9*log2e) ----
    // byte r of cm0 <-> key kb + 8*hi + r ; matches D-layout row M = r + 8*hi
    v8f s0, s1;
    {
      const uint32_t m0lo = (uint32_t)cm0, m0hi = (uint32_t)(cm0 >> 32);
      const uint32_t m1lo = (uint32_t)cm1, m1hi = (uint32_t)(cm1 >> 32);
#pragma unroll
      for (int r = 0; r < 4; ++r) {
        s0[r]     = (float)((m0lo >> (8 * r)) & 0xffu) * CMASK;  // v_cvt_f32_ubyteN
        s0[r + 4] = (float)((m0hi >> (8 * r)) & 0xffu) * CMASK;
        s1[r]     = (float)((m1lo >> (8 * r)) & 0xffu) * CMASK;
        s1[r + 4] = (float)((m1hi >> (8 * r)) & 0xffu) * CMASK;
      }
    }

    // ---- S^T = K x Q^T (+mask C) : A = K tile (lane = key, halves = d) ----
    const _Float16* kr0 = &ldsK[buf][ln][0];        // keys base+0..15
    const _Float16* kr1 = &ldsK[buf][16 + ln][0];   // keys base+16..31
    v16h a00 = cat16(ld8(kr0 + hi * 16),      ld8(kr0 + 8 + hi * 16));  // d 0..31
    v16h a01 = cat16(ld8(kr0 + 32 + hi * 16), ld8(kr0 + 40 + hi * 16));  // d 32..63
    v16h a10 = cat16(ld8(kr1 + hi * 16),      ld8(kr1 + 8 + hi * 16));
    v16h a11 = cat16(ld8(kr1 + 32 + hi * 16), ld8(kr1 + 40 + hi * 16));

    s0 = __builtin_amdgcn_wmma_f32_16x16x32_f16(false, a00, false, bq0, (short)0, s0, false, false);
    s1 = __builtin_amdgcn_wmma_f32_16x16x32_f16(false, a10, false, bq0, (short)0, s1, false, false);
    s0 = __builtin_amdgcn_wmma_f32_16x16x32_f16(false, a01, false, bq1, (short)0, s0, false, false);
    s1 = __builtin_amdgcn_wmma_f32_16x16x32_f16(false, a11, false, bq1, (short)0, s1, false, false);

    // ---- online softmax in exp2 domain (rows = queries = lanes) ----
    // balanced max tree (depth 4) instead of a serial chain
    float ma = fmaxf(fmaxf(s0[0], s0[1]), fmaxf(s0[2], s0[3]));
    float mb = fmaxf(fmaxf(s0[4], s0[5]), fmaxf(s0[6], s0[7]));
    float mc = fmaxf(fmaxf(s1[0], s1[1]), fmaxf(s1[2], s1[3]));
    float md = fmaxf(fmaxf(s1[4], s1[5]), fmaxf(s1[6], s1[7]));
    float mx = fmaxf(fmaxf(ma, mb), fmaxf(mc, md));
    mx = fmaxf(mx, xor16(mx));                    // join the two key-halves
    const float m_new = fmaxf(m_run, mx);
    const float alpha = __builtin_amdgcn_exp2f(m_run - m_new);
    float p0[8], p1[8];
#pragma unroll
    for (int r = 0; r < 8; ++r) {
      p0[r] = __builtin_amdgcn_exp2f(s0[r] - m_new);
      p1[r] = __builtin_amdgcn_exp2f(s1[r] - m_new);
    }
    // rescale only when some lane saw a new max (wave-uniform branch, EXEC intact)
    if (!__all(m_new == m_run)) {
#pragma unroll
      for (int r = 0; r < 8; ++r) {
        acc[0][r] *= alpha; acc[1][r] *= alpha; acc[2][r] *= alpha; acc[3][r] *= alpha;
      }
    }
    m_run = m_new;

    // ---- P^T -> B layout with permuted k-dim: every lane keeps its OWN data.
    // sigma swaps key groups 8..15 <-> 16..23; applied to both B (P^T) and A (V^T),
    // so bp[r] = p0[r], bp[8+r] = p1[r] for all lanes (no shuffles, no selects).
    v16h bp;
#pragma unroll
    for (int r = 0; r < 8; r += 2) {
      h2 t0 = pkrtz(p0[r], p0[r + 1]);
      h2 t1 = pkrtz(p1[r], p1[r + 1]);
      bp[r] = t0[0]; bp[r + 1] = t0[1];
      bp[r + 8] = t1[0]; bp[r + 9] = t1[1];
    }

    // ---- row-sum via ones-WMMA: every lane gets sum over all 32 block keys ----
    v8f ls = __builtin_amdgcn_wmma_f32_16x16x32_f16(false, aone, false, bp, (short)0, zz, false, false);

    // ---- O^T += V^T x P^T : A reads VT[d][hi*8 .. hi*8+7] and VT[d][16+hi*8 .. 16+hi*8+7] (sigma-consistent) ----
#pragma unroll
    for (int dt = 0; dt < 4; ++dt) {
      const _Float16* vr = &ldsVT[buf][dt * 16 + ln][hi * 8];
      v16h av = cat16(ld8(vr), ld8(vr + 16));
      acc[dt] = __builtin_amdgcn_wmma_f32_16x16x32_f16(false, av, false, bp, (short)0, acc[dt], false, false);
    }

    l_run = fmaf(l_run, alpha, ls[0]);
  }

  // ---- epilogue: O = acc / l ; O^T tile: lane = query, VGPR = d ----
  const float inv = 1.0f / l_run;
#pragma unroll
  for (int dt = 0; dt < 4; ++dt) {
#pragma unroll
    for (int r = 0; r < 8; ++r) {
      __builtin_nontemporal_store(acc[dt][r] * inv, &Orow[dt * 16 + r + 8 * hi]);
    }
  }
}

extern "C" void kernel_launch(void* const* d_in, const int* in_sizes, int n_in,
                              void* d_out, int out_size, void* d_ws, size_t ws_size,
                              hipStream_t stream) {
  const float* Q = (const float*)d_in[0];
  const float* K = (const float*)d_in[1];
  const float* V = (const float*)d_in[2];
  const unsigned char* M = (const unsigned char*)d_in[3];  // jax bool -> 1 byte
  float* O = (float*)d_out;

  const int BH = in_sizes[0] / (SEQ * DH);   // B*H = 64
  dim3 grid(SEQ / 64, BH, 1);                // 64 queries per 128-thread WG (4 waves)
  fa_fwd_kernel<<<grid, dim3(128, 1, 1), 0, stream>>>(Q, K, V, M, O);
}
